// AttentionModule_2911987827475
// MI455X (gfx1250) — matrix operand order repaired and checked by reading, not compiled, not run on hardware
//
#include <hip/hip_runtime.h>

typedef __attribute__((ext_vector_type(16))) _Float16 v16h;
typedef __attribute__((ext_vector_type(8)))  _Float16 v8h;
typedef __attribute__((ext_vector_type(8)))  float    v8f;
typedef __attribute__((ext_vector_type(4)))  int      v4i;

#if defined(__has_builtin)
#  if __has_builtin(__builtin_amdgcn_global_load_async_to_lds_b128)
#    define FA_HAS_ASYNC 1
#  endif
#endif
#ifndef FA_HAS_ASYNC
#  define FA_HAS_ASYNC 0
#endif

#define AS1 __attribute__((address_space(1)))
#define AS3 __attribute__((address_space(3)))

static constexpr int kB = 4;
static constexpr int kS = 4096;
static constexpr int kD = 64;
static constexpr int BM = 64;        // query rows per block
static constexpr int BN = 64;        // key rows per tile
static constexpr int NTHREADS = 128; // 4 waves (wave32)
static constexpr int STG  = 68;      // padded fp32 row stride, stage tiles
static constexpr int STRH = 72;      // padded f16 row stride, operand tiles

__device__ __forceinline__ void wait_async_zero() {
#if FA_HAS_ASYNC
#  if __has_builtin(__builtin_amdgcn_s_wait_asynccnt)
  __builtin_amdgcn_s_wait_asynccnt(0);
#  else
  asm volatile("s_wait_asynccnt 0x0" ::: "memory");
#  endif
#endif
}

__device__ __forceinline__ void async_b128(const float* g, float* l) {
#if FA_HAS_ASYNC
  __builtin_amdgcn_global_load_async_to_lds_b128((AS1 v4i*)g, (AS3 v4i*)l, 0, 0);
#else
  *(float4*)l = *(const float4*)g;
#endif
}

__device__ __forceinline__ v8f wmma_f16(v16h a, v16h b, v8f c) {
  return __builtin_amdgcn_wmma_f32_16x16x32_f16(false, a, false, b, (short)0, c,
                                                false, false);
}

__global__ __launch_bounds__(NTHREADS) void fa_fwd_cdna5(
    const float* __restrict__ q, const float* __restrict__ k,
    const float* __restrict__ v, float* __restrict__ o) {
  __shared__ float     sk[BN][STG];        // fp32 K stage (async dest)
  __shared__ float     sv[BN][STG];        // fp32 V stage (async dest)
  __shared__ _Float16  lkh[BN][STRH];      // K tile, f16 row-major
  __shared__ _Float16  lvT[kD][STRH];      // V tile, f16 transposed [d][key]
  __shared__ _Float16  lp[NTHREADS / 32][16][STRH];  // P staging per wave

  const int tid  = threadIdx.x;
  const int w    = tid >> 5;
  const int lane = tid & 31;
  const int ln   = lane & 15;
  const int hi   = lane >> 4;

  const int qb = blockIdx.x * BM;
  const int b  = blockIdx.y;

  const float L2E = 1.4426950408889634f;

  // ---- Q tile in A-operand layout, 1/sqrt(D)=0.125 folded in (exact) ----
  const int qrowA = qb + w * 16 + ln;
  const float* qp = q + ((size_t)b * kS + qrowA) * kD;
  v16h aq0, aq1;
#pragma unroll
  for (int e = 0; e < 8; ++e) {
    aq0[e]     = (_Float16)(qp[hi * 8 + e] * 0.125f);
    aq0[e + 8] = (_Float16)(qp[hi * 8 + 16 + e] * 0.125f);
    aq1[e]     = (_Float16)(qp[32 + hi * 8 + e] * 0.125f);
    aq1[e + 8] = (_Float16)(qp[32 + hi * 8 + 16 + e] * 0.125f);
  }

  v16h bones;                       // all-ones B tile: row-sum via WMMA
#pragma unroll
  for (int e = 0; e < 16; ++e) bones[e] = (_Float16)1.0f;

  v8f vzero;
#pragma unroll
  for (int i = 0; i < 8; ++i) vzero[i] = 0.0f;

  float rmax[8];
  v8f acco[5];                      // [0..3]=O tile 16x64, [4]=row-sum column
#pragma unroll
  for (int r = 0; r < 8; ++r) rmax[r] = -1e30f;
#pragma unroll
  for (int dt = 0; dt < 5; ++dt) acco[dt] = vzero;

  const float* kbase = k + (size_t)b * kS * kD;
  const float* vbase = v + (size_t)b * kS * kD;
  const int nkb = qb / BN + 1;      // causal: tiles 0 .. diag

  // 64 rows x 16 chunks(16B) per tile; 8 K-chunks + 8 V-chunks per thread
  auto issue_tile = [&](int jj) {
    const int k0 = jj * BN;
#pragma unroll
    for (int t = 0; t < 8; ++t) {
      const int idx = t * NTHREADS + tid;   // 0..1023
      const int rr = idx >> 4, cc = idx & 15;
      async_b128(kbase + (size_t)(k0 + rr) * kD + cc * 4, &sk[rr][cc * 4]);
      async_b128(vbase + (size_t)(k0 + rr) * kD + cc * 4, &sv[rr][cc * 4]);
    }
  };

  issue_tile(0);
  wait_async_zero();
  __syncthreads();

  for (int j = 0; j < nkb; ++j) {
    const int k0 = j * BN;

    // ---- cooperative one-shot fp32 -> f16 conversion ----
    // K: thread -> (row rr, 32-col half hh), row-major f16
    {
      const int rr = tid >> 1;
      const int hh = (tid & 1) * 32;
#pragma unroll
      for (int m = 0; m < 4; ++m) {
        float4 f0 = *(const float4*)&sk[rr][hh + m * 8 + 0];
        float4 f1 = *(const float4*)&sk[rr][hh + m * 8 + 4];
        v8h h;
        h[0] = (_Float16)f0.x; h[1] = (_Float16)f0.y;
        h[2] = (_Float16)f0.z; h[3] = (_Float16)f0.w;
        h[4] = (_Float16)f1.x; h[5] = (_Float16)f1.y;
        h[6] = (_Float16)f1.z; h[7] = (_Float16)f1.w;
        *(v8h*)&lkh[rr][hh + m * 8] = h;
      }
    }
    // V: thread -> (key pair kp, swizzled 16-wide d group), transposed f16
    {
      const int kp = tid & 31;
      const int dg = tid >> 5;
      const int d0 = ((dg + (kp >> 3)) & 3) * 16;
#pragma unroll
      for (int m = 0; m < 4; ++m) {
        float4 a0 = *(const float4*)&sv[2 * kp + 0][d0 + m * 4];
        float4 a1 = *(const float4*)&sv[2 * kp + 1][d0 + m * 4];
        lvT[d0 + m * 4 + 0][2 * kp] = (_Float16)a0.x;
        lvT[d0 + m * 4 + 0][2 * kp + 1] = (_Float16)a1.x;
        lvT[d0 + m * 4 + 1][2 * kp] = (_Float16)a0.y;
        lvT[d0 + m * 4 + 1][2 * kp + 1] = (_Float16)a1.y;
        lvT[d0 + m * 4 + 2][2 * kp] = (_Float16)a0.z;
        lvT[d0 + m * 4 + 2][2 * kp + 1] = (_Float16)a1.z;
        lvT[d0 + m * 4 + 3][2 * kp] = (_Float16)a0.w;
        lvT[d0 + m * 4 + 3][2 * kp + 1] = (_Float16)a1.w;
      }
    }
    __syncthreads();                 // stage consumed + f16 buffers ready

    if (j + 1 < nkb) issue_tile(j + 1);   // DMA overlaps compute below

    // ---- scores S = (Qs) K^T : batch-load operands, then WMMA chains ----
    v8f sc[4];
    {
      v16h bk[4];
#pragma unroll
      for (int nt = 0; nt < 4; ++nt) {            // low-K half (d 0..31)
        const _Float16* kr = &lkh[nt * 16 + ln][hi * 8];
#pragma unroll
        for (int e = 0; e < 8; ++e) { bk[nt][e] = kr[e]; bk[nt][e + 8] = kr[16 + e]; }
      }
#pragma unroll
      for (int nt = 0; nt < 4; ++nt) sc[nt] = wmma_f16(aq0, bk[nt], vzero);
#pragma unroll
      for (int nt = 0; nt < 4; ++nt) {            // high-K half (d 32..63)
        const _Float16* kr = &lkh[nt * 16 + ln][32 + hi * 8];
#pragma unroll
        for (int e = 0; e < 8; ++e) { bk[nt][e] = kr[e]; bk[nt][e + 8] = kr[16 + e]; }
      }
#pragma unroll
      for (int nt = 0; nt < 4; ++nt) sc[nt] = wmma_f16(aq1, bk[nt], sc[nt]);
    }

    // ---- causal mask: only the diagonal tile needs it (uniform branch) ----
    if (j == nkb - 1) {
#pragma unroll
      for (int nt = 0; nt < 4; ++nt) {
        const int col = k0 + nt * 16 + ln;
#pragma unroll
        for (int r = 0; r < 8; ++r) {
          const int row = qb + w * 16 + r + 8 * hi;
          if (col > row) sc[nt][r] = -1e30f;
        }
      }
    }

    // ---- online softmax: max-reduce + rescale; row sums come from WMMA ----
#pragma unroll
    for (int r = 0; r < 8; ++r) {
      float t = fmaxf(fmaxf(sc[0][r], sc[1][r]), fmaxf(sc[2][r], sc[3][r]));
      t = fmaxf(t, __shfl_xor(t, 1, 16));
      t = fmaxf(t, __shfl_xor(t, 2, 16));
      t = fmaxf(t, __shfl_xor(t, 4, 16));
      t = fmaxf(t, __shfl_xor(t, 8, 16));
      const float mnew = fmaxf(rmax[r], t);
      const float corr = __builtin_amdgcn_exp2f((rmax[r] - mnew) * L2E);
      rmax[r] = mnew;
      sc[0][r] = __builtin_amdgcn_exp2f((sc[0][r] - mnew) * L2E);
      sc[1][r] = __builtin_amdgcn_exp2f((sc[1][r] - mnew) * L2E);
      sc[2][r] = __builtin_amdgcn_exp2f((sc[2][r] - mnew) * L2E);
      sc[3][r] = __builtin_amdgcn_exp2f((sc[3][r] - mnew) * L2E);
#pragma unroll
      for (int dt = 0; dt < 5; ++dt) acco[dt][r] *= corr;
    }

    // ---- P: C layout -> f16 A layout via per-wave LDS staging ----
#pragma unroll
    for (int nt = 0; nt < 4; ++nt)
#pragma unroll
      for (int r = 0; r < 8; ++r)
        lp[w][r + 8 * hi][nt * 16 + ln] = (_Float16)sc[nt][r];

    asm volatile("s_wait_dscnt 0x0" ::: "memory");  // wave-local LDS RAW

    // ---- O += P V (8 WMMAs) and rowsum += P 1s (2 WMMAs) ----
#pragma unroll
    for (int kk = 0; kk < 2; ++kk) {
      v16h ap, bv[4];
      const _Float16* pr = &lp[w][ln][kk * 32 + hi * 8];
#pragma unroll
      for (int e = 0; e < 8; ++e) { ap[e] = pr[e]; ap[e + 8] = pr[16 + e]; }
#pragma unroll
      for (int dt = 0; dt < 4; ++dt) {             // batch all B operands
        const _Float16* vr = &lvT[dt * 16 + ln][kk * 32 + hi * 8];
#pragma unroll
        for (int e = 0; e < 8; ++e) { bv[dt][e] = vr[e]; bv[dt][e + 8] = vr[16 + e]; }
      }
#pragma unroll
      for (int dt = 0; dt < 4; ++dt)
        acco[dt] = wmma_f16(ap, bv[dt], acco[dt]);
      acco[4] = wmma_f16(ap, bones, acco[4]);      // row sums, all lanes
    }

    wait_async_zero();   // next stage tiles landed (wave-local)
    __syncthreads();     // all waves: compute done, stage j+1 visible
  }

  // ---- epilogue: normalize by WMMA-computed row sums and store ----
#pragma unroll
  for (int r = 0; r < 8; ++r) {
    const float inv = 1.0f / acco[4][r];
    const int row = qb + w * 16 + r + 8 * hi;
    float* op = o + ((size_t)b * kS + row) * kD;
#pragma unroll
    for (int dt = 0; dt < 4; ++dt)
      op[dt * 16 + ln] = acco[dt][r] * inv;
  }
}

extern "C" void kernel_launch(void* const* d_in, const int* in_sizes, int n_in,
                              void* d_out, int out_size, void* d_ws, size_t ws_size,
                              hipStream_t stream) {
  const float* q = (const float*)d_in[0];
  const float* k = (const float*)d_in[1];
  const float* v = (const float*)d_in[2];
  float* o = (float*)d_out;
  dim3 grid(kS / BM, kB, 1);
  dim3 block(NTHREADS, 1, 1);
  fa_fwd_cdna5<<<grid, block, 0, stream>>>(q, k, v, o);
}
